// Decoder_29137058136065
// MI455X (gfx1250) — compile-verified, matrix operand order checked by reading, not run on hardware
//
#include <hip/hip_runtime.h>
#include <hip/hip_bf16.h>

#define B_   2
#define N_   4096
#define D_   512
#define H_   8
#define DK_  64
#define BLK_ 128
#define TB_  32
#define W_   8

typedef __attribute__((ext_vector_type(16))) _Float16 v16h;
typedef __attribute__((ext_vector_type(8)))  _Float16 v8h;
typedef __attribute__((ext_vector_type(4)))  _Float16 v4h;
typedef __attribute__((ext_vector_type(8)))  float    v8f;
typedef __attribute__((ext_vector_type(4)))  float    v4f;

struct F8 { union { v8f v; float f[8]; }; };

__device__ __forceinline__ v8f vzero() { v8f z = {}; return z; }

// Load one 16x32 f16 WMMA operand fragment from a row-major matrix whose K
// dimension is contiguous (used for A, and for B in NT-form GEMMs).
// ISA layout (CDNA5 ISA): lane L (0..15) = row L, halves 0..7 ->
// K = hi*8..hi*8+7, halves 8..15 -> K = 16+hi*8.., where hi = lane>>4.
__device__ __forceinline__ v16h load_frag(const _Float16* base, int ld) {
  const int lane = threadIdx.x & 31;
  const _Float16* p = base + (size_t)(lane & 15) * (size_t)ld + ((lane >> 4) << 3);
  v8h lo = *(const v8h*)(p);
  v8h hi = *(const v8h*)(p + 16);
  return __builtin_shufflevector(lo, hi, 0,1,2,3,4,5,6,7,8,9,10,11,12,13,14,15);
}

__device__ __forceinline__ v8f wmma_f16(v16h a, v16h b, v8f c) {
  return __builtin_amdgcn_wmma_f32_16x16x32_f16(false, a, false, b, (short)0, c,
                                                false, false);
}

// gfx1250 async copy: 16 bytes global -> LDS, tracked by ASYNCcnt.
// VDST VGPR carries the workgroup-relative LDS byte offset (low 32 bits of the
// generic shared address); VADDR is the 64-bit global address.
__device__ __forceinline__ void async_g2l_b128(const _Float16* lds_dst,
                                               const _Float16* gsrc) {
  const unsigned loff = (unsigned)(uintptr_t)lds_dst;
  const unsigned long long gaddr = (unsigned long long)(uintptr_t)gsrc;
  asm volatile("global_load_async_to_lds_b128 %0, %1, off"
               :: "v"(loff), "v"(gaddr) : "memory");
}
__device__ __forceinline__ void wait_async_le8() {
  asm volatile("s_wait_asynccnt 0x8" ::: "memory");
}
__device__ __forceinline__ void wait_async_0() {
  asm volatile("s_wait_asynccnt 0x0" ::: "memory");
}

// ---------------------------------------------------------------------------
// fp32 -> f16 conversion, vectorized (b128 load, b64 store)
// ---------------------------------------------------------------------------
__global__ void convert_f16_kernel(const float* __restrict__ in,
                                   _Float16* __restrict__ out, int n4) {
  int idx = blockIdx.x * blockDim.x + threadIdx.x;
  int stride = gridDim.x * blockDim.x;
  for (; idx < n4; idx += stride) {
    v4f x = ((const v4f*)in)[idx];
    v4h y;
    #pragma unroll
    for (int j = 0; j < 4; ++j) y[j] = (_Float16)x[j];
    ((v4h*)out)[idx] = y;
  }
}

// ---------------------------------------------------------------------------
// Fused QKV projection:  Y = x @ W^T + b  for W in {Wq, Wk, Wv}
// Q,K stored row-major f16 (B*N, D); V stored transposed per head:
// vt[b][h][dk][n]  (f16) so that P*V later is an NT GEMM.
// Block: 256 thr (8 waves, 4x2), block tile 128(M) x 64(N), wave tile 32x32.
// ---------------------------------------------------------------------------
__global__ void __launch_bounds__(256)
qkv_proj_kernel(const _Float16* __restrict__ xh,
                const _Float16* __restrict__ wq,
                const _Float16* __restrict__ wk,
                const _Float16* __restrict__ wv,
                const float* __restrict__ bq,
                const float* __restrict__ bk,
                const float* __restrict__ bv,
                _Float16* __restrict__ qh,
                _Float16* __restrict__ kh,
                _Float16* __restrict__ vt) {
  const int w  = threadIdx.x >> 5;
  const int wm = w & 3, wn = w >> 2;
  const int m0 = blockIdx.x * 128 + wm * 32;
  const int n0 = blockIdx.y * 64  + wn * 32;
  const int which = blockIdx.z;
  const _Float16* wt   = (which == 0) ? wq : ((which == 1) ? wk : wv);
  const float*    bias = (which == 0) ? bq : ((which == 1) ? bk : bv);

  F8 acc[2][2];
  #pragma unroll
  for (int i = 0; i < 2; ++i)
    #pragma unroll
    for (int j = 0; j < 2; ++j) acc[i][j].v = vzero();

  for (int k = 0; k < D_; k += 32) {
    v16h a[2], b[2];
    #pragma unroll
    for (int i = 0; i < 2; ++i)
      a[i] = load_frag(xh + (size_t)(m0 + i * 16) * D_ + k, D_);
    #pragma unroll
    for (int j = 0; j < 2; ++j)
      b[j] = load_frag(wt + (size_t)(n0 + j * 16) * D_ + k, D_);
    #pragma unroll
    for (int i = 0; i < 2; ++i)
      #pragma unroll
      for (int j = 0; j < 2; ++j)
        acc[i][j].v = wmma_f16(a[i], b[j], acc[i][j].v);
  }

  const int lane = threadIdx.x & 31;
  const int ln = lane & 15, hif = lane >> 4;
  #pragma unroll
  for (int i = 0; i < 2; ++i) {
    #pragma unroll
    for (int j = 0; j < 2; ++j) {
      const int col   = n0 + j * 16 + ln;
      const float bb_ = bias[col];
      const int rbase = m0 + i * 16 + 8 * hif;
      if (which < 2) {
        _Float16* outp = (which == 0) ? qh : kh;
        #pragma unroll
        for (int v = 0; v < 8; ++v)
          outp[(size_t)(rbase + v) * D_ + col] = (_Float16)(acc[i][j].f[v] + bb_);
      } else {
        const int hh = col >> 6, dk = col & 63;
        const int bI = rbase >> 12;
        const int nn = rbase & (N_ - 1);
        v8h pk;
        #pragma unroll
        for (int v = 0; v < 8; ++v) pk[v] = (_Float16)(acc[i][j].f[v] + bb_);
        *(v8h*)(vt + ((size_t)(bI * H_ + hh) * DK_ + dk) * N_ + nn) = pk;
      }
    }
  }
}

// ---------------------------------------------------------------------------
// Block-sparse causal flash attention with double-buffered async K/V staging.
// One workgroup per (b, h, q-block i). 8 waves; wave w owns 16 q rows.
// K tile (128x64) and transposed-V tile (64x128) are staged Global->LDS with
// GLOBAL_LOAD_ASYNC_TO_LDS_B128 (ASYNCcnt), shared by all 8 waves.
// ---------------------------------------------------------------------------
__global__ void __launch_bounds__(256)
attn_kernel(const _Float16* __restrict__ qh,
            const _Float16* __restrict__ kh,
            const _Float16* __restrict__ vt,
            _Float16* __restrict__ ah) {
  __shared__ __align__(16) _Float16 Ks[2][BLK_][DK_];   // 2 x 16 KB
  __shared__ __align__(16) _Float16 Vs[2][DK_][BLK_];   // 2 x 16 KB
  __shared__ __align__(16) _Float16 Pl[8][16 * BLK_];   // 32 KB

  const int blk = blockIdx.x;
  const int i   = blk & (TB_ - 1);
  const int bh  = blk >> 5;
  const int h   = bh & (H_ - 1);
  const int bb  = bh >> 3;
  const int tid  = threadIdx.x;
  const int w    = tid >> 5;
  const int lane = tid & 31;
  const int ln = lane & 15, hif = lane >> 4;

  const int qrow = i * BLK_ + w * 16;
  const _Float16* qbase = qh + ((size_t)(bb * N_ + qrow)) * D_ + h * DK_;
  const _Float16* kbase = kh + (size_t)bb * N_ * D_ + h * DK_;
  const _Float16* vbase = vt + ((size_t)(bb * H_ + h)) * DK_ * N_;

  // ---- async staging of one window block's K and V tiles ----
  // K tile: 128 rows x 64 halves  = 1024 16B chunks; 4 per thread.
  // V tile:  64 rows x 128 halves = 1024 16B chunks; 4 per thread.
  auto stage_kv = [&](int buf, int jb) {
    #pragma unroll
    for (int it = 0; it < 4; ++it) {
      const int c = tid + it * 256;
      const int r = c >> 3, col = (c & 7) * 8;       // 8 chunks per K row
      async_g2l_b128(&Ks[buf][r][col],
                     kbase + (size_t)(jb * BLK_ + r) * D_ + col);
    }
    #pragma unroll
    for (int it = 0; it < 4; ++it) {
      const int c = tid + it * 256;
      const int r = c >> 4, col = (c & 15) * 8;      // 16 chunks per V row
      async_g2l_b128(&Vs[buf][r][col],
                     vbase + (size_t)r * N_ + jb * BLK_ + col);
    }
  };

  // Q fragments are loop-invariant
  v16h aq[2];
  #pragma unroll
  for (int k2 = 0; k2 < 2; ++k2) aq[k2] = load_frag(qbase + k2 * 32, D_);

  F8 o[4];
  float mrow[8], lrow[8];
  #pragma unroll
  for (int u = 0; u < 4; ++u) o[u].v = vzero();
  #pragma unroll
  for (int v = 0; v < 8; ++v) { mrow[v] = -__builtin_inff(); lrow[v] = 0.f; }

  const float scale = 0.125f;  // TAU / sqrt(DK)
  const int j0 = (i >= W_ - 1) ? (i - (W_ - 1)) : 0;

  stage_kv(0, j0);  // prologue

  for (int jb = j0; jb <= i; ++jb) {
    const int cur = (jb - j0) & 1;
    const bool pf = (jb + 1 <= i);       // block-uniform
    if (pf) stage_kv(cur ^ 1, jb + 1);   // prefetch next window block
    if (pf) wait_async_le8();            // current buffer's 8 chunks done
    else    wait_async_0();
    __syncthreads();                     // all waves' slices visible

    // ---- S = (Q K^T) * scale, 16 x 128 for this key block ----
    F8 s[8];
    #pragma unroll
    for (int t = 0; t < 8; ++t) {
      s[t].v = vzero();
      #pragma unroll
      for (int k2 = 0; k2 < 2; ++k2) {
        v16h bk = load_frag(&Ks[cur][t * 16][k2 * 32], DK_);
        s[t].v = wmma_f16(aq[k2], bk, s[t].v);
      }
    }
    // causal mask + scale
    #pragma unroll
    for (int t = 0; t < 8; ++t) {
      const int kpos = jb * BLK_ + t * 16 + ln;
      #pragma unroll
      for (int v = 0; v < 8; ++v) {
        const int qpos = qrow + 8 * hif + v;
        const float sv = s[t].f[v] * scale;
        s[t].f[v] = (kpos <= qpos) ? sv : -__builtin_inff();
      }
    }
    // ---- online softmax ----
    float fac[8];
    #pragma unroll
    for (int v = 0; v < 8; ++v) {
      float mj = s[0].f[v];
      #pragma unroll
      for (int t = 1; t < 8; ++t) mj = fmaxf(mj, s[t].f[v]);
      #pragma unroll
      for (int d = 1; d < 16; d <<= 1) mj = fmaxf(mj, __shfl_xor(mj, d, 32));
      const float mnew = fmaxf(mrow[v], mj);
      fac[v]  = __expf(mrow[v] - mnew);
      mrow[v] = mnew;
    }
    #pragma unroll
    for (int v = 0; v < 8; ++v) {
      float rs = 0.f;
      #pragma unroll
      for (int t = 0; t < 8; ++t) {
        const float p = __expf(s[t].f[v] - mrow[v]);
        s[t].f[v] = p;
        rs += p;
      }
      #pragma unroll
      for (int d = 1; d < 16; d <<= 1) rs += __shfl_xor(rs, d, 32);
      lrow[v] = lrow[v] * fac[v] + rs;
      #pragma unroll
      for (int u = 0; u < 4; ++u) o[u].f[v] *= fac[v];
    }
    // ---- P (f16) -> wave-private LDS tile (C layout -> row-major) ----
    #pragma unroll
    for (int t = 0; t < 8; ++t)
      #pragma unroll
      for (int v = 0; v < 8; ++v)
        Pl[w][(8 * hif + v) * BLK_ + t * 16 + ln] = (_Float16)s[t].f[v];
    __syncthreads();
    // ---- O += P V  (NT GEMM against staged transposed V) ----
    #pragma unroll
    for (int kc = 0; kc < 4; ++kc) {
      v16h ap = load_frag(&Pl[w][kc * 32], BLK_);
      #pragma unroll
      for (int u = 0; u < 4; ++u) {
        v16h bv = load_frag(&Vs[cur][u * 16][kc * 32], BLK_);
        o[u].v = wmma_f16(ap, bv, o[u].v);
      }
    }
    __syncthreads();  // done reading `cur` before it is re-staged
  }

  float inv[8];
  #pragma unroll
  for (int v = 0; v < 8; ++v) inv[v] = 1.f / lrow[v];
  #pragma unroll
  for (int u = 0; u < 4; ++u) {
    const int dcol = h * DK_ + u * 16 + ln;
    #pragma unroll
    for (int v = 0; v < 8; ++v)
      ah[(size_t)(bb * N_ + qrow + 8 * hif + v) * D_ + dcol] =
          (_Float16)(o[u].f[v] * inv[v]);
  }
}

// ---------------------------------------------------------------------------
// Output projection: out = attn @ Wo^T + bo  (fp32 output)
// ---------------------------------------------------------------------------
__global__ void __launch_bounds__(256)
out_proj_kernel(const _Float16* __restrict__ ahp,
                const _Float16* __restrict__ wo,
                const float* __restrict__ bo,
                float* __restrict__ out) {
  const int w  = threadIdx.x >> 5;
  const int wm = w & 3, wn = w >> 2;
  const int m0 = blockIdx.x * 128 + wm * 32;
  const int n0 = blockIdx.y * 64  + wn * 32;

  F8 acc[2][2];
  #pragma unroll
  for (int i = 0; i < 2; ++i)
    #pragma unroll
    for (int j = 0; j < 2; ++j) acc[i][j].v = vzero();

  for (int k = 0; k < D_; k += 32) {
    v16h a[2], b[2];
    #pragma unroll
    for (int i = 0; i < 2; ++i)
      a[i] = load_frag(ahp + (size_t)(m0 + i * 16) * D_ + k, D_);
    #pragma unroll
    for (int j = 0; j < 2; ++j)
      b[j] = load_frag(wo + (size_t)(n0 + j * 16) * D_ + k, D_);
    #pragma unroll
    for (int i = 0; i < 2; ++i)
      #pragma unroll
      for (int j = 0; j < 2; ++j)
        acc[i][j].v = wmma_f16(a[i], b[j], acc[i][j].v);
  }

  const int lane = threadIdx.x & 31;
  const int ln = lane & 15, hif = lane >> 4;
  #pragma unroll
  for (int i = 0; i < 2; ++i) {
    #pragma unroll
    for (int j = 0; j < 2; ++j) {
      const int col   = n0 + j * 16 + ln;
      const float bb_ = bo[col];
      const int rbase = m0 + i * 16 + 8 * hif;
      #pragma unroll
      for (int v = 0; v < 8; ++v)
        out[(size_t)(rbase + v) * D_ + col] = acc[i][j].f[v] + bb_;
    }
  }
}

// ---------------------------------------------------------------------------
extern "C" void kernel_launch(void* const* d_in, const int* in_sizes, int n_in,
                              void* d_out, int out_size, void* d_ws, size_t ws_size,
                              hipStream_t stream) {
  const float* x    = (const float*)d_in[0];
  const float* wq_w = (const float*)d_in[1];
  const float* wq_b = (const float*)d_in[2];
  const float* wk_w = (const float*)d_in[3];
  const float* wk_b = (const float*)d_in[4];
  const float* wv_w = (const float*)d_in[5];
  const float* wv_b = (const float*)d_in[6];
  const float* wo_w = (const float*)d_in[7];
  const float* wo_b = (const float*)d_in[8];
  float* out = (float*)d_out;

  const size_t BN = (size_t)B_ * N_;
  const size_t XE = BN * D_;
  const size_t WE = (size_t)D_ * D_;

  char* ws = (char*)d_ws;
  size_t off = 0;
  auto alloc = [&](size_t bytes) -> void* {
    void* p = ws + off;
    off += (bytes + 255) & ~(size_t)255;
    return p;
  };
  _Float16* xh  = (_Float16*)alloc(XE * 2);
  _Float16* wqh = (_Float16*)alloc(WE * 2);
  _Float16* wkh = (_Float16*)alloc(WE * 2);
  _Float16* wvh = (_Float16*)alloc(WE * 2);
  _Float16* woh = (_Float16*)alloc(WE * 2);
  _Float16* qh  = (_Float16*)alloc(XE * 2);
  _Float16* kh  = (_Float16*)alloc(XE * 2);
  _Float16* vt  = (_Float16*)alloc(XE * 2);  // [B][H][DK][N]
  _Float16* ah  = (_Float16*)alloc(XE * 2);

  convert_f16_kernel<<<1024, 256, 0, stream>>>(x,    xh,  (int)(XE / 4));
  convert_f16_kernel<<<256,  256, 0, stream>>>(wq_w, wqh, (int)(WE / 4));
  convert_f16_kernel<<<256,  256, 0, stream>>>(wk_w, wkh, (int)(WE / 4));
  convert_f16_kernel<<<256,  256, 0, stream>>>(wv_w, wvh, (int)(WE / 4));
  convert_f16_kernel<<<256,  256, 0, stream>>>(wo_w, woh, (int)(WE / 4));

  dim3 gp((unsigned)(BN / 128), D_ / 64, 3);
  qkv_proj_kernel<<<gp, 256, 0, stream>>>(xh, wqh, wkh, wvh, wq_b, wk_b, wv_b,
                                          qh, kh, vt);

  attn_kernel<<<B_ * H_ * TB_, 256, 0, stream>>>(qh, kh, vt, ah);

  dim3 go((unsigned)(BN / 128), D_ / 64);
  out_proj_kernel<<<go, 256, 0, stream>>>(ah, woh, wo_b, out);
}
